// FullCrossAttention_16441134809855
// MI455X (gfx1250) — matrix operand order repaired and checked by reading, not compiled, not run on hardware
//
#include <hip/hip_runtime.h>
#include <hip/hip_bf16.h>

// ---------------- types ----------------
typedef __bf16 v16bf __attribute__((ext_vector_type(16)));
typedef float  v8f   __attribute__((ext_vector_type(8)));
typedef unsigned int u32x4 __attribute__((ext_vector_type(4)));
typedef int i32x8 __attribute__((ext_vector_type(8)));
typedef int i32x4 __attribute__((ext_vector_type(4)));

#define B_   16
#define SEQ_ 256
#define DIM_ 1024
#define HEADS_ 16
#define DHEAD_ 64
#define KV_  2125
#define MS_  2048
#define PS_  16
#define SCALE_ 0.125f
#define NEG_INF_ (-1e30f)

union Frag { v16bf v; uint4 u[2]; };

static __device__ inline v16bf ld_frag(const __bf16* p0, const __bf16* p1) {
    Frag f;
    f.u[0] = *(const uint4*)p0;
    f.u[1] = *(const uint4*)p1;
    return f.v;
}

static __device__ inline v8f vzero8() {
    v8f z = {0.f,0.f,0.f,0.f,0.f,0.f,0.f,0.f};
    return z;
}

// ---------------- cast fp32 -> bf16 ----------------
__global__ void cast_f32_bf16(const float* __restrict__ in, __bf16* __restrict__ out, int n) {
    int i = blockIdx.x * 256 + threadIdx.x;
    if (i < n) out[i] = (__bf16)in[i];
}

// ---------------- tiled WMMA GEMM ----------------
// C[M,1024] = A[M,1024](bf16) * W[1024,1024](bf16)
// mode 0: fp32 out [M][N], optional bias[N]
// mode 1: bf16 out permuted to [b][h][j][64] with rowsPerBatch = Rper
__global__ __launch_bounds__(256)
void gemm_bf16(const __bf16* __restrict__ A, const __bf16* __restrict__ W,
               void* __restrict__ Out, const float* __restrict__ bias,
               int Mrows, int mode, int Rper)
{
    const int K = 1024, N = 1024;
    const int LDA = 40; // 32 + 8 pad (bf16 elems) -> conflict-free b128 reads
    __shared__ __attribute__((aligned(16))) __bf16 lA[128 * 40]; // [row][k]
    __shared__ __attribute__((aligned(16))) __bf16 lB[128 * 40]; // [n][k] (transposed)

    const int tid  = threadIdx.x;
    const int lane = tid & 31;
    const int wave = tid >> 5;
    const int wm   = wave >> 1;   // 0..3 -> rows wm*32
    const int wn   = wave & 1;    // 0..1 -> cols wn*64
    const int m0   = blockIdx.y * 128;
    const int n0   = blockIdx.x * 128;

    const int mr    = lane & 15;
    const int abase = (lane < 16) ? 0 : 8;   // A frag K offsets: {abase, abase+16}
    const int bbase = (lane < 16) ? 0 : 8;   // B frag K offsets: {bbase, bbase+16}

    v8f acc[2][4];
#pragma unroll
    for (int mt = 0; mt < 2; ++mt)
#pragma unroll
        for (int n = 0; n < 4; ++n) acc[mt][n] = vzero8();

    const int arow = tid >> 1, acol = (tid & 1) * 16;         // A: 128 rows x 32 k
    const int brow = tid >> 3, bcol = (tid & 7) * 16;         // W: 32 k-rows x 128 n

    for (int k0 = 0; k0 < K; k0 += 32) {
        // stage A tile
        {
            uint4 d0 = {0,0,0,0}, d1 = {0,0,0,0};
            int gm = m0 + arow;
            if (gm < Mrows) {
                const uint4* src = (const uint4*)(A + (size_t)gm * K + k0 + acol);
                d0 = src[0]; d1 = src[1];
                if (k0 + 32 < K) __builtin_prefetch(A + (size_t)gm * K + k0 + 32 + acol, 0, 3);
            }
            uint4* dst = (uint4*)(lA + arow * LDA + acol);
            dst[0] = d0; dst[1] = d1;
        }
        // stage W tile, transposed to [n][k]
        {
            const __bf16* src = W + (size_t)(k0 + brow) * N + n0 + bcol;
            Frag d;
            d.u[0] = ((const uint4*)src)[0];
            d.u[1] = ((const uint4*)src)[1];
            if (k0 + 32 < K) __builtin_prefetch(src + 32 * N, 0, 3);
            __bf16 tmp[16];
            *(Frag*)tmp = d;
#pragma unroll
            for (int q = 0; q < 16; ++q)
                lB[(bcol + q) * LDA + brow] = tmp[q];
        }
        __syncthreads();

        v16bf afr[2];
#pragma unroll
        for (int mt = 0; mt < 2; ++mt) {
            const __bf16* ap = lA + (wm * 32 + mt * 16 + mr) * LDA + abase;
            afr[mt] = ld_frag(ap, ap + 16);
        }
        v16bf bfr[4];
#pragma unroll
        for (int n = 0; n < 4; ++n) {
            const __bf16* bp = lB + (wn * 64 + n * 16 + mr) * LDA + bbase;
            bfr[n] = ld_frag(bp, bp + 16);
        }
#pragma unroll
        for (int mt = 0; mt < 2; ++mt)
#pragma unroll
            for (int n = 0; n < 4; ++n)
                acc[mt][n] = __builtin_amdgcn_wmma_f32_16x16x32_bf16(
                    false, afr[mt], false, bfr[n], (short)0, acc[mt][n], false, false);
        __syncthreads();
    }

    // writeback
    const int rbase = (lane < 16) ? 0 : 8;
#pragma unroll
    for (int mt = 0; mt < 2; ++mt) {
#pragma unroll
        for (int n = 0; n < 4; ++n) {
#pragma unroll
            for (int r = 0; r < 8; ++r) {
                int row = m0 + wm * 32 + mt * 16 + rbase + r;
                int col = n0 + wn * 64 + n * 16 + mr;
                if (row < Mrows) {
                    float val = acc[mt][n][r];
                    if (mode == 0) {
                        if (bias) val += bias[col];
                        ((float*)Out)[(size_t)row * N + col] = val;
                    } else {
                        int b = row / Rper, j = row - b * Rper;
                        int h = col >> 6, d = col & 63;
                        ((__bf16*)Out)[((((size_t)b * HEADS_ + h) * Rper) + j) * DHEAD_ + d] = (__bf16)val;
                    }
                }
            }
        }
    }
}

// ---------------- rotary on Q (in place, bf16) ----------------
__global__ void rotary_q_kernel(__bf16* __restrict__ Qb, const float* __restrict__ freqs,
                                const int* __restrict__ cur)
{
    int idx = blockIdx.x * 256 + threadIdx.x;        // B*H*SEQ*32
    if (idx >= B_ * HEADS_ * SEQ_ * 32) return;
    int d2 = idx & 31; int t = idx >> 5;
    int s  = t & 255;

    int cp = cur[0];
    int inner = cp & 15;
    int rs = inner & 3, cs = inner >> 2;
    int fr = rs * PS_ + (s >> 4), fc = cs * PS_ + (s & 15);
    const float* f = freqs + ((size_t)fr * 64 + fc) * DHEAD_ + 2 * d2;
    __bf16* p = Qb + (size_t)(idx >> 5) * DHEAD_ + 2 * d2;
    float q0 = (float)p[0], q1 = (float)p[1];
    float f0 = f[0], f1 = f[1];
    float o0 = q0 * __cosf(f0) - q1 * __sinf(f0);
    float o1 = q1 * __cosf(f1) + q0 * __sinf(f1);
    p[0] = (__bf16)o0; p[1] = (__bf16)o1;
}

// ---------------- bias + rotary on memory part of K (in place, bf16) ----------------
__global__ void biasrot_k_kernel(__bf16* __restrict__ Kb, const float* __restrict__ freqs,
                                 const float* __restrict__ rel_table,
                                 const int* __restrict__ mem_idxs, const int* __restrict__ cur)
{
    int idx = blockIdx.x * 256 + threadIdx.x;        // B*H*MS*32
    if (idx >= B_ * HEADS_ * MS_ * 32) return;
    int d2 = idx & 31; int t = idx >> 5;
    int j  = t & (MS_ - 1); t >>= 11;
    int h  = t & 15;  int b = t >> 4;

    int m = j >> 8;        // mem slot
    int i = j & 255;       // pos within patch

    int cp = cur[0];
    int inner = cp & 15;
    int rs = inner & 3, cs = inner >> 2, ts = cp >> 4;
    int mi = mem_idxs[m];
    int in2 = mi & 15;
    int rt = in2 & 3, ct = in2 >> 2, tt = mi >> 4;
    int rel = (rt - rs + 4) * 9 + (ct - cs + 4) + (tt - ts + 4) * 81;
    float bias = rel_table[rel * HEADS_ + h];

    int fr = rt * PS_ + (i >> 4), fc = ct * PS_ + (i & 15);
    const float* f = freqs + ((size_t)fr * 64 + fc) * DHEAD_ + 2 * d2;
    __bf16* p = Kb + (((((size_t)b * HEADS_ + h) * KV_) + j) * DHEAD_) + 2 * d2;
    float k0 = (float)p[0] + bias, k1 = (float)p[1] + bias;
    float f0 = f[0], f1 = f[1];
    float o0 = k0 * __cosf(f0) - k1 * __sinf(f0);
    float o1 = k1 * __cosf(f1) + k0 * __sinf(f1);
    p[0] = (__bf16)o0; p[1] = (__bf16)o1;
}

// ---------------- flash attention: one block per (b,h) ----------------
__global__ __launch_bounds__(256)
void attn_kernel(const __bf16* __restrict__ Q, const __bf16* __restrict__ Kc,
                 const __bf16* __restrict__ Vc, __bf16* __restrict__ O)
{
    const int LDK = 72; // 64 + 8 pad (matches TDM pad_interval=32dw, pad_amount=4dw)
    const int LDV = 40; // 32 + 8 pad
    __shared__ __attribute__((aligned(16))) __bf16 lK[32 * 72];       // [kv][dim]
    __shared__ __attribute__((aligned(16))) __bf16 lV[64 * 40];       // [dim][kv] transposed
    __shared__ __attribute__((aligned(16))) __bf16 lP[8 * 16 * 40];   // per-wave P tile

    const int bh = blockIdx.x;                 // b*16 + h
    const int b  = bh >> 4, h = bh & 15;
    const int tid = threadIdx.x, wave = tid >> 5, lane = tid & 31;

    const __bf16* Qbh = Q  + (size_t)bh * SEQ_ * DHEAD_;
    const __bf16* Kbh = Kc + (size_t)bh * KV_  * DHEAD_;
    const __bf16* Vbh = Vc + (size_t)bh * KV_  * DHEAD_;

    const int mr    = lane & 15;
    const int abase = (lane < 16) ? 0 : 8;
    const int bbase = (lane < 16) ? 0 : 8;

    // Q fragments for this wave's 32 rows: [mt][kstep]
    v16bf qf[2][2];
#pragma unroll
    for (int mt = 0; mt < 2; ++mt)
#pragma unroll
        for (int ks = 0; ks < 2; ++ks) {
            const __bf16* qp = Qbh + (size_t)(wave * 32 + mt * 16 + mr) * DHEAD_ + ks * 32 + abase;
            qf[mt][ks] = ld_frag(qp, qp + 16);
        }

    v8f om[2][4];
    float mrow[2][8], lrow[2][8];
#pragma unroll
    for (int mt = 0; mt < 2; ++mt) {
#pragma unroll
        for (int nd = 0; nd < 4; ++nd) om[mt][nd] = vzero8();
#pragma unroll
        for (int r = 0; r < 8; ++r) { mrow[mt][r] = NEG_INF_; lrow[mt][r] = 0.f; }
    }

    const int krow = tid >> 3, kcol = (tid & 7) * 8;   // 32 kv x 64 dim cooperative

    for (int j0 = 0; j0 < KV_; j0 += 32) {
        // ---- stage V chunk cooperatively (transposed, zero-padded) ----
        {
            int j = j0 + krow;
            Frag e; e.u[0] = (uint4){0,0,0,0};
            if (j < KV_) e.u[0] = *(const uint4*)(Vbh + (size_t)j * DHEAD_ + kcol);
            __bf16 tmp[8];
            *(uint4*)tmp = e.u[0];
#pragma unroll
            for (int q = 0; q < 8; ++q)
                lV[(kcol + q) * LDV + krow] = tmp[q];
        }

        // ---- stage K chunk: TDM tensor DMA if available, else cooperative ----
#if __has_builtin(__builtin_amdgcn_tensor_load_to_lds)
        if (wave == 0) {
            // D# group0: count=1, lds_addr, 57-bit global addr, type=2 (image)
            unsigned ldsK = (unsigned)(size_t)(&lK[0]);           // low 32 bits = LDS byte offset
            unsigned long long ga = (unsigned long long)(size_t)(Kbh + (size_t)j0 * DHEAD_);
            unsigned rem = (unsigned)(KV_ - j0);                  // rows left; OOB rows read as 0
            u32x4 g0;
            g0[0] = 1u;                                           // count=1 (valid descriptor)
            g0[1] = ldsK;
            g0[2] = (unsigned)(ga & 0xFFFFFFFFu);
            g0[3] = (unsigned)((ga >> 32) & 0x01FFFFFFu) | (2u << 30);
            // D# group1: data_size=2B; pad: every 32 DWORDs (128B row) add 4 DWORDs (16B) -> LDK=72
            i32x8 g1;
            g1[0] = (1 << 16) | (1 << 20) | (4 << 22) | (3 << 25);
            g1[1] = (int)(64u << 16);                             // tensor_dim0 = 64 (bits 79:48 lo)
            g1[2] = (int)((rem & 0xFFFFu) << 16);                 // tensor_dim1 lo (bits 111:80)
            g1[3] = (int)((rem >> 16) | (64u << 16));             // tensor_dim1 hi | tile_dim0=64
            g1[4] = 32;                                           // tile_dim1=32, tile_dim2=0
            g1[5] = 64;                                           // tensor_dim0_stride = 64
            g1[6] = 0;
            g1[7] = 0;
            i32x4 z4 = {0,0,0,0};
            i32x8 z8 = {0,0,0,0,0,0,0,0};
            __builtin_amdgcn_tensor_load_to_lds(g0, g1, z4, z4, z8, 0);
            __builtin_amdgcn_s_wait_tensorcnt(0);
        }
#else
        {
            int j = j0 + krow;
            Frag d; d.u[0] = (uint4){0,0,0,0};
            if (j < KV_) d.u[0] = *(const uint4*)(Kbh + (size_t)j * DHEAD_ + kcol);
            *(uint4*)(lK + krow * LDK + kcol) = d.u[0];
        }
#endif
        __syncthreads();

        // K^T fragments [n][ks], V fragments [nd] (shared by both mt)
        v16bf kf[2][2];
#pragma unroll
        for (int n = 0; n < 2; ++n)
#pragma unroll
            for (int ks = 0; ks < 2; ++ks) {
                const __bf16* kp = lK + (n * 16 + mr) * LDK + ks * 32 + bbase;
                kf[n][ks] = ld_frag(kp, kp + 16);
            }
        v16bf vf[4];
#pragma unroll
        for (int nd = 0; nd < 4; ++nd) {
            const __bf16* vp = lV + (nd * 16 + mr) * LDV + bbase;
            vf[nd] = ld_frag(vp, vp + 16);
        }

#pragma unroll
        for (int mt = 0; mt < 2; ++mt) {
            v8f s[2] = { vzero8(), vzero8() };
#pragma unroll
            for (int n = 0; n < 2; ++n)
#pragma unroll
                for (int ks = 0; ks < 2; ++ks)
                    s[n] = __builtin_amdgcn_wmma_f32_16x16x32_bf16(
                        false, qf[mt][ks], false, kf[n][ks], (short)0, s[n], false, false);

            // scale + mask out-of-range kv columns
#pragma unroll
            for (int n = 0; n < 2; ++n) {
                bool oob = (j0 + n * 16 + mr) >= KV_;
#pragma unroll
                for (int r = 0; r < 8; ++r)
                    s[n][r] = oob ? NEG_INF_ : s[n][r] * SCALE_;
            }

            // online softmax: row max over 32 columns (16 lanes per half)
            float cmax[8];
#pragma unroll
            for (int r = 0; r < 8; ++r) cmax[r] = fmaxf(s[0][r], s[1][r]);
#pragma unroll
            for (int dmask = 1; dmask < 16; dmask <<= 1)
#pragma unroll
                for (int r = 0; r < 8; ++r)
                    cmax[r] = fmaxf(cmax[r], __shfl_xor(cmax[r], dmask, 32));

            float alpha[8];
#pragma unroll
            for (int r = 0; r < 8; ++r) {
                float mn = fmaxf(mrow[mt][r], cmax[r]);
                alpha[r] = __expf(mrow[mt][r] - mn);
                mrow[mt][r] = mn;
            }
#pragma unroll
            for (int nd = 0; nd < 4; ++nd)
#pragma unroll
                for (int r = 0; r < 8; ++r) om[mt][nd][r] *= alpha[r];

            float rsum[8];
#pragma unroll
            for (int r = 0; r < 8; ++r) rsum[r] = 0.f;
#pragma unroll
            for (int n = 0; n < 2; ++n)
#pragma unroll
                for (int r = 0; r < 8; ++r) {
                    float p = __expf(s[n][r] - mrow[mt][r]);
                    s[n][r] = p;
                    rsum[r] += p;
                }
#pragma unroll
            for (int dmask = 1; dmask < 16; dmask <<= 1)
#pragma unroll
                for (int r = 0; r < 8; ++r)
                    rsum[r] += __shfl_xor(rsum[r], dmask, 32);
#pragma unroll
            for (int r = 0; r < 8; ++r)
                lrow[mt][r] = lrow[mt][r] * alpha[r] + rsum[r];

            // C-layout P -> LDS -> A-layout fragment (same-wave round trip)
            __bf16* pl = lP + wave * (16 * LDV);
            int prow = (lane < 16) ? 0 : 8;
#pragma unroll
            for (int n = 0; n < 2; ++n)
#pragma unroll
                for (int r = 0; r < 8; ++r)
                    pl[(prow + r) * LDV + n * 16 + mr] = (__bf16)s[n][r];
            asm volatile("s_wait_dscnt 0" ::: "memory");
            const __bf16* pp = pl + mr * LDV + abase;
            v16bf pf = ld_frag(pp, pp + 16);

            // P @ V
#pragma unroll
            for (int nd = 0; nd < 4; ++nd)
                om[mt][nd] = __builtin_amdgcn_wmma_f32_16x16x32_bf16(
                    false, pf, false, vf[nd], (short)0, om[mt][nd], false, false);
        }
        __syncthreads();
    }

    // write O[b][q][h][d] (bf16), normalized
    const int rbase = (lane < 16) ? 0 : 8;
#pragma unroll
    for (int mt = 0; mt < 2; ++mt) {
#pragma unroll
        for (int nd = 0; nd < 4; ++nd) {
#pragma unroll
            for (int r = 0; r < 8; ++r) {
                int q = wave * 32 + mt * 16 + rbase + r;
                int d = nd * 16 + mr;
                float val = om[mt][nd][r] / lrow[mt][r];
                O[(((size_t)b * SEQ_ + q) * HEADS_ + h) * DHEAD_ + d] = (__bf16)val;
            }
        }
    }
}

// ---------------- host side ----------------
extern "C" void kernel_launch(void* const* d_in, const int* in_sizes, int n_in,
                              void* d_out, int out_size, void* d_ws, size_t ws_size,
                              hipStream_t stream)
{
    const float* x    = (const float*)d_in[0];
    const float* cond = (const float*)d_in[1];
    const float* freq = (const float*)d_in[2];
    const float* Wq   = (const float*)d_in[3];
    const float* Wk   = (const float*)d_in[4];
    const float* Wv   = (const float*)d_in[5];
    const float* rel  = (const float*)d_in[6];
    const float* Wo   = (const float*)d_in[7];
    const float* bo   = (const float*)d_in[8];
    const int* mem    = (const int*)d_in[9];
    const int* cur    = (const int*)d_in[10];

    const size_t Mq = (size_t)B_ * SEQ_;        // 4096
    const size_t Mk = (size_t)B_ * KV_;         // 34000
    char* ws = (char*)d_ws;
    size_t off = 0;
    auto alloc = [&](size_t bytes) { char* p = ws + off; off += (bytes + 255) & ~(size_t)255; return p; };

    __bf16* xb    = (__bf16*)alloc(Mq * DIM_ * 2);
    __bf16* condb = (__bf16*)alloc(Mk * DIM_ * 2);
    __bf16* Wqb   = (__bf16*)alloc((size_t)DIM_ * DIM_ * 2);
    __bf16* Wkb   = (__bf16*)alloc((size_t)DIM_ * DIM_ * 2);
    __bf16* Wvb   = (__bf16*)alloc((size_t)DIM_ * DIM_ * 2);
    __bf16* Wob   = (__bf16*)alloc((size_t)DIM_ * DIM_ * 2);
    __bf16* Qb    = (__bf16*)alloc(Mq * DIM_ * 2);   // [b][h][s][64]
    __bf16* Kb    = (__bf16*)alloc(Mk * DIM_ * 2);   // [b][h][j][64]
    __bf16* Vb    = (__bf16*)alloc(Mk * DIM_ * 2);   // [b][h][j][64]
    __bf16* Ob    = (__bf16*)alloc(Mq * DIM_ * 2);   // [b][s][h][64]

    auto cdiv = [](size_t a, size_t b) { return (int)((a + b - 1) / b); };

    cast_f32_bf16<<<cdiv(Mq * DIM_, 256), 256, 0, stream>>>(x, xb, (int)(Mq * DIM_));
    cast_f32_bf16<<<cdiv(Mk * DIM_, 256), 256, 0, stream>>>(cond, condb, (int)(Mk * DIM_));
    cast_f32_bf16<<<cdiv((size_t)DIM_ * DIM_, 256), 256, 0, stream>>>(Wq, Wqb, DIM_ * DIM_);
    cast_f32_bf16<<<cdiv((size_t)DIM_ * DIM_, 256), 256, 0, stream>>>(Wk, Wkb, DIM_ * DIM_);
    cast_f32_bf16<<<cdiv((size_t)DIM_ * DIM_, 256), 256, 0, stream>>>(Wv, Wvb, DIM_ * DIM_);
    cast_f32_bf16<<<cdiv((size_t)DIM_ * DIM_, 256), 256, 0, stream>>>(Wo, Wob, DIM_ * DIM_);

    dim3 gq(8, cdiv(Mq, 128));
    gemm_bf16<<<gq, 256, 0, stream>>>(xb, Wqb, Qb, nullptr, (int)Mq, 1, SEQ_);
    dim3 gk(8, cdiv(Mk, 128));
    gemm_bf16<<<gk, 256, 0, stream>>>(condb, Wkb, Kb, nullptr, (int)Mk, 1, KV_);
    gemm_bf16<<<gk, 256, 0, stream>>>(condb, Wvb, Vb, nullptr, (int)Mk, 1, KV_);

    rotary_q_kernel<<<cdiv((size_t)B_ * HEADS_ * SEQ_ * 32, 256), 256, 0, stream>>>(Qb, freq, cur);
    biasrot_k_kernel<<<cdiv((size_t)B_ * HEADS_ * MS_ * 32, 256), 256, 0, stream>>>(Kb, freq, rel, mem, cur);

    attn_kernel<<<B_ * HEADS_, 256, 0, stream>>>(Qb, Kb, Vb, Ob);

    gemm_bf16<<<gq, 256, 0, stream>>>(Ob, Wob, d_out, bo, (int)Mq, 0, 0);
}
